// SwinBlock_37314675868126
// MI455X (gfx1250) — matrix operand order repaired and checked by reading, not compiled, not run on hardware
//
#include <hip/hip_runtime.h>
#include <hip/hip_bf16.h>

typedef __bf16 bf16;
typedef __attribute__((ext_vector_type(16))) __bf16 v16bf;
typedef __attribute__((ext_vector_type(8)))  float  v8f;
typedef unsigned int u32;
typedef __attribute__((ext_vector_type(4))) u32 v4u;
typedef __attribute__((ext_vector_type(4))) int v4i;
typedef __attribute__((ext_vector_type(8))) int v8i;

#define BATCH 128
#define CCH   512
#define HH    14
#define WWID  14
#define WIN   7
#define SHIFT 3
#define HEADS 16
#define HD    32
#define HID   2048
#define NTOK  49
#define NWIN  4
#define MROWS (BATCH*NWIN*NTOK)   // 25088 = 16*1568 = 128*196

// ---------------------------------------------------------------------------
// TDM: 1-D copy of `nbytes` (multiple of 8) from global to LDS. Issued once
// per calling wave; completion tracked with TENSORcnt.
// D# layout per CDNA5 ISA 8.3/8.4 (group0: count/lds/global/type,
// group1: data_size=8B, tensor_dim0=tile_dim0=elems, dims1=1).
// ---------------------------------------------------------------------------
__device__ __forceinline__ void tdm_load_1d(u32 lds_byte_addr, const void* gsrc, u32 nbytes) {
    u32 elems = nbytes >> 3;                       // 8-byte elements
    unsigned long long ga = (unsigned long long)(size_t)gsrc;
    v4u g0;
    g0[0] = 1u;                                    // count=1 (valid), no gather
    g0[1] = lds_byte_addr;                         // lds_addr (bits 63:32)
    g0[2] = (u32)ga;                               // global_addr[31:0]
    g0[3] = ((u32)(ga >> 32) & 0x01ffffffu) | (2u << 30);  // addr[56:32] | type=2
    v8i g1;
    g1[0] = (int)(3u << 16);                       // wg_mask=0, data_size=3 (8B)
    g1[1] = (int)((elems & 0xffffu) << 16);        // tensor_dim0[15:0]  @63:48
    g1[2] = (int)(((elems >> 16) & 0xffffu) | (1u << 16)); // dim0[31:16]@79:64, dim1=1
    g1[3] = (int)((elems & 0xffffu) << 16);        // tile_dim0 @127:112
    g1[4] = 1;                                     // tile_dim1=1 @143:128
    g1[5] = (int)elems;                            // tensor_dim0_stride[31:0]
    g1[6] = 0;
    g1[7] = 0;
    v4i z = {0, 0, 0, 0};
#if __clang_major__ >= 23
    v8i z8 = {0, 0, 0, 0, 0, 0, 0, 0};
    __builtin_amdgcn_tensor_load_to_lds(g0, g1, z, z, z8, 0);
#else
    __builtin_amdgcn_tensor_load_to_lds(g0, g1, z, z, 0);
#endif
}

// ---------------------------------------------------------------------------
// LayerNorm over channels + (mode 0) shift/window-partition from NCHW input,
// or (mode 1) plain NHWC row emit from NCHW residual. One wave per pixel.
// ---------------------------------------------------------------------------
__global__ void ln_kernel(const float* __restrict__ src, const float* __restrict__ w,
                          const float* __restrict__ b, bf16* __restrict__ dst, int mode) {
    int wave = (blockIdx.x * blockDim.x + threadIdx.x) >> 5;
    int lane = threadIdx.x & 31;
    if (wave >= MROWS) return;

    int bIdx, hs, ws_;
    if (mode == 0) {
        int bw = wave / NTOK, n = wave % NTOK;
        bIdx = bw >> 2; int wdw = bw & 3;
        int wi = wdw >> 1, wj = wdw & 1;
        int pi = n / WIN, pj = n % WIN;
        hs  = (wi*WIN + pi + SHIFT) % HH;     // roll(-SHIFT) => read ahead
        ws_ = (wj*WIN + pj + SHIFT) % WWID;
    } else {
        bIdx = wave / (HH*WWID);
        int p = wave % (HH*WWID);
        hs = p / WWID; ws_ = p % WWID;
    }
    const float* base = src + (size_t)bIdx * CCH * (HH*WWID) + hs*WWID + ws_;

    float vals[16];
    float s = 0.f, ss = 0.f;
#pragma unroll
    for (int j = 0; j < 16; ++j) {
        float v = base[(size_t)(lane + 32*j) * (HH*WWID)];
        vals[j] = v; s += v; ss += v*v;
    }
#pragma unroll
    for (int off = 16; off >= 1; off >>= 1) {
        s  += __shfl_xor(s,  off, 32);
        ss += __shfl_xor(ss, off, 32);
    }
    float mean = s * (1.f/CCH);
    float var  = ss * (1.f/CCH) - mean*mean;
    float rstd = rsqrtf(var + 1e-5f);

    bf16* out = dst + (size_t)wave * CCH;
#pragma unroll
    for (int j = 0; j < 16; ++j) {
        int c = lane + 32*j;
        out[c] = (bf16)((vals[j] - mean) * rstd * w[c] + b[c]);
    }
}

// ---------------------------------------------------------------------------
// Pack a (Nout x K) fp32 weight (row-major, y = x @ W^T) into a contiguous-k
// bf16 B-fragment order for v_wmma_f32_16x16x32_bf16:
//   packed[((kblk*nTilesN + tileN)*32 + lane)*16 + i]
//     = W[tileN*16 + lane%16][kblk*32 + (lane/16)*16 + i]
// so each lane reads its 16 halves as one contiguous 32-byte chunk.
// ---------------------------------------------------------------------------
__global__ void pack_w(const float* __restrict__ W, bf16* __restrict__ out, int K, int Nout) {
    int t = blockIdx.x * blockDim.x + threadIdx.x;
    if (t >= K * Nout) return;
    int i    = t & 15;
    int lane = (t >> 4) & 31;
    int rest = t >> 9;
    int nTiles = Nout >> 4;
    int tileN  = rest % nTiles;
    int kblk   = rest / nTiles;
    int n = tileN*16 + (lane & 15);
    int k = kblk*32 + ((lane >> 4) << 4) + i;
    out[t] = (bf16)W[(size_t)n * K + k];
}

// ---------------------------------------------------------------------------
// WMMA bf16 GEMM, block tile 128x64: 8 waves x (1 tileM, 4 tileN).
// B panel (4KB per K-step) staged into LDS by wave0 via TDM, double-buffered,
// consumed by all waves via ds_load_b128. A streamed from global.
// epi 0: store f32 (ld=N)                              -> outF
// epi 1: proj: window-reverse + roll + gamma*val+resid -> NCHW outF
// epi 2: GELU(exact) -> bf16 outB (ld=N)
// epi 3: fc2: NHWC row -> NCHW, + gamma*val + resid    -> outF
// Grid must be exactly (M/128)*(N/64) blocks of 256 threads.
// ---------------------------------------------------------------------------
union ABfrag { uint4 u[2]; v16bf v; };

__global__ void gemm_wmma(const bf16* __restrict__ A, const bf16* __restrict__ Bp,
                          const float* __restrict__ bias,
                          int M, int N, int K, int epi,
                          float* __restrict__ outF, bf16* __restrict__ outB,
                          const float* __restrict__ resid, const float* __restrict__ gamma) {
    __shared__ __align__(16) bf16 bpanel[2][4 * 512];   // 2 x 4KB double buffer

    int lane   = threadIdx.x & 31;
    int waveId = threadIdx.x >> 5;
    int nBlkN  = N >> 6;
    int nb     = blockIdx.x % nBlkN;       // 64-col group
    int mb     = blockIdx.x / nBlkN;       // 128-row group
    int nTilesN = N >> 4;

    int tileM  = mb*8 + waveId;
    int row    = tileM*16 + (lane & 15);
    int laneHi = lane >> 4;

    v8f acc[4];
#pragma unroll
    for (int j = 0; j < 4; ++j) {
        float bv = bias[nb*64 + j*16 + (lane & 15)];
#pragma unroll
        for (int i = 0; i < 8; ++i) acc[j][i] = bv;
    }

    const bf16* aRow = A + (size_t)row * K + laneHi * 16;
    // global packed-B source for this block's 4 consecutive N-tiles
    const bf16* bSrc = Bp + ((size_t)nb * 4) * 512;
    const size_t bKStrideElems = (size_t)nTilesN * 512;       // per K-step

    u32 lds[2];
    lds[0] = (u32)(size_t)&bpanel[0][0];
    lds[1] = (u32)(size_t)&bpanel[1][0];

    bool w0 = (waveId == 0);
    int nk = K >> 5;

    if (w0) tdm_load_1d(lds[0], bSrc, 4096);                  // prime kb=0

    for (int kb = 0; kb < nk; ++kb) {
        if (w0) {
            if (kb + 1 < nk) {
                tdm_load_1d(lds[(kb + 1) & 1], bSrc + (size_t)(kb + 1) * bKStrideElems, 4096);
                __builtin_amdgcn_s_wait_tensorcnt(1);         // kb's panel landed
            } else {
                __builtin_amdgcn_s_wait_tensorcnt(0);
            }
        }
        __syncthreads();                                      // panel kb visible to all

        ABfrag a;
        const uint4* ap = (const uint4*)(aRow + kb*32);
        a.u[0] = ap[0];          // K halves 0..7   (hi lanes: 16..23)
        a.u[1] = ap[1];          // K halves 8..15  (hi lanes: 24..31)
        if (kb + 1 < nk) __builtin_prefetch(aRow + (kb+1)*32, 0, 1);

        const bf16* bl = &bpanel[kb & 1][lane * 16];
#pragma unroll
        for (int j = 0; j < 4; ++j) {
            ABfrag b;
            const uint4* bp = (const uint4*)(bl + j * 512);
            b.u[0] = bp[0];
            b.u[1] = bp[1];
            acc[j] = __builtin_amdgcn_wmma_f32_16x16x32_bf16(
                         false, a.v, false, b.v, (short)0, acc[j], false, false);
        }
        __syncthreads();                                      // done with panel kb
    }

    if (epi == 0) {
#pragma unroll
        for (int j = 0; j < 4; ++j) {
            int col = nb*64 + j*16 + (lane & 15);
#pragma unroll
            for (int i = 0; i < 8; ++i) {
                int m = tileM*16 + i + laneHi*8;
                outF[(size_t)m * N + col] = acc[j][i];
            }
        }
    } else if (epi == 1) {
#pragma unroll
        for (int j = 0; j < 4; ++j) {
            int col = nb*64 + j*16 + (lane & 15);
#pragma unroll
            for (int i = 0; i < 8; ++i) {
                int r = tileM*16 + i + laneHi*8;
                int bw = r / NTOK, n = r % NTOK;
                int bb = bw >> 2, wdw = bw & 3;
                int wi = wdw >> 1, wj = wdw & 1;
                int hh = wi*WIN + n/WIN, ww2 = wj*WIN + n%WIN;
                int h  = hh + SHIFT;  if (h  >= HH)   h  -= HH;   // roll(+SHIFT)
                int w2 = ww2 + SHIFT; if (w2 >= WWID) w2 -= WWID;
                size_t idx = ((size_t)bb*CCH + col)*(HH*WWID) + h*WWID + w2;
                outF[idx] = resid[idx] + gamma[col] * acc[j][i];
            }
        }
    } else if (epi == 2) {
#pragma unroll
        for (int j = 0; j < 4; ++j) {
            int col = nb*64 + j*16 + (lane & 15);
#pragma unroll
            for (int i = 0; i < 8; ++i) {
                int m = tileM*16 + i + laneHi*8;
                float v = acc[j][i];
                v = 0.5f * v * (1.f + erff(v * 0.70710678118654752f));
                outB[(size_t)m * N + col] = (bf16)v;
            }
        }
    } else {
#pragma unroll
        for (int j = 0; j < 4; ++j) {
            int col = nb*64 + j*16 + (lane & 15);
#pragma unroll
            for (int i = 0; i < 8; ++i) {
                int r = tileM*16 + i + laneHi*8;
                int bb = r / (HH*WWID), p = r % (HH*WWID);
                size_t idx = ((size_t)bb*CCH + col)*(HH*WWID) + p;
                outF[idx] = resid[idx] + gamma[col] * acc[j][i];
            }
        }
    }
}

// ---------------------------------------------------------------------------
// Window attention core: one block per (batch-window, head). 49x49 scores in
// LDS, f32 softmax with relative bias + shift mask, output bf16 rows.
// (~3% of total FLOPs -> VALU path is roofline-appropriate.)
// ---------------------------------------------------------------------------
__global__ void attn_kernel(const float* __restrict__ qkv, const float* __restrict__ relb,
                            bf16* __restrict__ out) {
    int bh = blockIdx.x;
    int bw = bh / HEADS, head = bh % HEADS;
    int wdw = bw & 3;
    int wi = wdw >> 1, wj = wdw & 1;

    __shared__ float kls[NTOK*HD];
    __shared__ float vls[NTOK*HD];
    __shared__ float sls[NTOK*NTOK];

    int tid = threadIdx.x;
    const float* base = qkv + (size_t)bw * NTOK * (3*CCH);
    for (int idx = tid; idx < NTOK*HD; idx += blockDim.x) {
        int m = idx >> 5, d = idx & 31;
        kls[idx] = base[(size_t)m*(3*CCH) + CCH   + head*HD + d];
        vls[idx] = base[(size_t)m*(3*CCH) + 2*CCH + head*HD + d];
    }
    __syncthreads();

    int n = tid;
    if (n < NTOK) {
        float q[HD];
        const float* qp = base + (size_t)n*(3*CCH) + head*HD;
#pragma unroll
        for (int d = 0; d < HD; ++d) q[d] = qp[d] * 0.17677669529663689f; // hd^-0.5

        int pi = n / WIN, pj = n % WIN;
        int gy = wi*WIN + pi, gx = wj*WIN + pj;
        int lbl_n = ((gy < WIN) ? 0 : ((gy < HH - SHIFT) ? 1 : 2)) * 3
                  + ((gx < WIN) ? 0 : ((gx < WWID - SHIFT) ? 1 : 2));

        float* srow = sls + n*NTOK;
        float mx = -1e30f;
        for (int m = 0; m < NTOK; ++m) {
            float s = 0.f;
#pragma unroll
            for (int d = 0; d < HD; ++d) s += q[d] * kls[m*HD + d];
            int mi = m / WIN, mj = m % WIN;
            int gy2 = wi*WIN + mi, gx2 = wj*WIN + mj;
            int lbl_m = ((gy2 < WIN) ? 0 : ((gy2 < HH - SHIFT) ? 1 : 2)) * 3
                      + ((gx2 < WIN) ? 0 : ((gx2 < WWID - SHIFT) ? 1 : 2));
            int ridx = (pi - mi + WIN - 1)*(2*WIN - 1) + (pj - mj + WIN - 1);
            s += relb[ridx*HEADS + head];
            if (lbl_m != lbl_n) s -= 100.f;
            srow[m] = s;
            mx = fmaxf(mx, s);
        }
        float sum = 0.f;
        for (int m = 0; m < NTOK; ++m) { float e = expf(srow[m] - mx); srow[m] = e; sum += e; }
        float inv = 1.f / sum;

        bf16* op = out + ((size_t)bw*NTOK + n)*CCH + head*HD;
#pragma unroll
        for (int d = 0; d < HD; ++d) {
            float o = 0.f;
            for (int m = 0; m < NTOK; ++m) o += srow[m] * vls[m*HD + d];
            op[d] = (bf16)(o * inv);
        }
    }
}

// ---------------------------------------------------------------------------
extern "C" void kernel_launch(void* const* d_in, const int* in_sizes, int n_in,
                              void* d_out, int out_size, void* d_ws, size_t ws_size,
                              hipStream_t stream) {
    const float* x      = (const float*)d_in[0];
    const float* ln1_w  = (const float*)d_in[1];
    const float* ln1_b  = (const float*)d_in[2];
    const float* qkv_w  = (const float*)d_in[3];
    const float* qkv_b  = (const float*)d_in[4];
    const float* relb   = (const float*)d_in[5];
    const float* proj_w = (const float*)d_in[6];
    const float* proj_b = (const float*)d_in[7];
    const float* gamma1 = (const float*)d_in[8];
    const float* ln2_w  = (const float*)d_in[9];
    const float* ln2_b  = (const float*)d_in[10];
    const float* fc1_w  = (const float*)d_in[11];
    const float* fc1_b  = (const float*)d_in[12];
    const float* fc2_w  = (const float*)d_in[13];
    const float* fc2_b  = (const float*)d_in[14];
    const float* gamma2 = (const float*)d_in[15];

    char* ws = (char*)d_ws;
    size_t off = 0;
    auto alloc = [&](size_t bytes) -> void* {
        void* p = ws + off;
        off = (off + bytes + 255) & ~(size_t)255;
        return p;
    };
    bf16*  xw     = (bf16*) alloc((size_t)MROWS * CCH * 2);
    float* qkvbuf = (float*)alloc((size_t)MROWS * 3*CCH * 4);
    bf16*  aout   = (bf16*) alloc((size_t)MROWS * CCH * 2);
    float* xres   = (float*)alloc((size_t)MROWS * CCH * 4);
    bf16*  z      = (bf16*) alloc((size_t)MROWS * CCH * 2);
    bf16*  hbuf   = (bf16*) alloc((size_t)MROWS * HID * 2);
    bf16*  wqkv   = (bf16*) alloc((size_t)CCH * 3*CCH * 2);
    bf16*  wproj  = (bf16*) alloc((size_t)CCH * CCH * 2);
    bf16*  wfc1   = (bf16*) alloc((size_t)CCH * HID * 2);
    bf16*  wfc2   = (bf16*) alloc((size_t)HID * CCH * 2);

    // weight packing (bf16 B-fragment layout)
    pack_w<<<(CCH*3*CCH + 255)/256, 256, 0, stream>>>(qkv_w,  wqkv,  CCH, 3*CCH);
    pack_w<<<(CCH*CCH   + 255)/256, 256, 0, stream>>>(proj_w, wproj, CCH, CCH);
    pack_w<<<(CCH*HID   + 255)/256, 256, 0, stream>>>(fc1_w,  wfc1,  CCH, HID);
    pack_w<<<(HID*CCH   + 255)/256, 256, 0, stream>>>(fc2_w,  wfc2,  HID, CCH);

    // LN1 + shift + window partition -> bf16 rows
    ln_kernel<<<MROWS/8, 256, 0, stream>>>(x, ln1_w, ln1_b, xw, 0);

    // qkv GEMM: (25088 x 512) @ (512 x 1536)
    gemm_wmma<<<(MROWS/128) * (3*CCH/64), 256, 0, stream>>>(
        xw, wqkv, qkv_b, MROWS, 3*CCH, CCH, 0, qkvbuf, nullptr, nullptr, nullptr);

    // window attention (softmax core)
    attn_kernel<<<BATCH*NWIN*HEADS, 64, 0, stream>>>(qkvbuf, relb, aout);

    // proj GEMM + window-reverse + roll + gamma1 + residual -> xres (NCHW)
    gemm_wmma<<<(MROWS/128) * (CCH/64), 256, 0, stream>>>(
        aout, wproj, proj_b, MROWS, CCH, CCH, 1, xres, nullptr, x, gamma1);

    // LN2 -> bf16 NHWC rows
    ln_kernel<<<MROWS/8, 256, 0, stream>>>(xres, ln2_w, ln2_b, z, 1);

    // fc1 GEMM + GELU -> bf16 hidden
    gemm_wmma<<<(MROWS/128) * (HID/64), 256, 0, stream>>>(
        z, wfc1, fc1_b, MROWS, HID, CCH, 2, nullptr, hbuf, nullptr, nullptr);

    // fc2 GEMM + gamma2 + residual -> d_out (NCHW f32)
    gemm_wmma<<<(MROWS/128) * (CCH/64), 256, 0, stream>>>(
        hbuf, wfc2, fc2_b, MROWS, CCH, HID, 3, (float*)d_out, nullptr, xres, gamma2);
}
